// MultiheadAttention_13365938225757
// MI455X (gfx1250) — matrix operand order repaired and checked by reading, not compiled, not run on hardware
//
#include <hip/hip_runtime.h>
#include <hip/hip_bf16.h>

typedef __attribute__((ext_vector_type(16))) __bf16 v16bf;
typedef __attribute__((ext_vector_type(8)))  __bf16 v8bf;
typedef __attribute__((ext_vector_type(4)))  __bf16 v4bf;
typedef __attribute__((ext_vector_type(8)))  float  v8f;

#define WMMA_BF16(A_, B_, C_) \
  __builtin_amdgcn_wmma_f32_16x16x32_bf16(false, (A_), false, (B_), (short)0, (C_), false, false)

constexpr int Td = 1024;   // queries per batch
constexpr int Sd = 1024;   // keys per batch
constexpr int Ad = 512;    // feature dim
constexpr int BQ = 32;     // query rows per workgroup
constexpr int BS = 64;     // S-block (keys per iteration)
constexpr int AC = 128;    // A-chunk staged per QK^T step / V chunk width

constexpr int SQ_LD  = 520; // bf16 elems (1040B row: 16B aligned, banks spread)
constexpr int SK_LD  = 136; // bf16 elems (272B row), K chunk stored [n][k]
constexpr int SVT_LD = 72;  // bf16 elems (144B row), V chunk stored transposed [n][k]
constexpr int SS_LD  = 66;  // f32 elems; cols 64/65 reused as softmax partial-max scratch
constexpr int SP_LD  = 72;  // bf16 elems (144B row, 16B aligned)

constexpr float kScale = 0.04419417382415922f; // 1/sqrt(512)

// A-fragment 16x32 bf16 (MxK). lane: m = mrow + (lane&15)
// lane<16: K runs [kb..kb+7],[kb+16..kb+23]; lane>=16: +8
__device__ __forceinline__ v16bf load_afrag(const __bf16* base, int ld, int mrow, int kbase, int lane) {
  const __bf16* p = base + (mrow + (lane & 15)) * ld + kbase + ((lane & 16) ? 8 : 0);
  v8bf lo = *reinterpret_cast<const v8bf*>(p);
  v8bf hi = *reinterpret_cast<const v8bf*>(p + 16);
  v16bf a;
#pragma unroll
  for (int e = 0; e < 8; ++e) { a[e] = lo[e]; a[8 + e] = hi[e]; }
  return a;
}

// B-fragment 32x16 bf16 (KxN) from tile stored [n][k] row-major.
// lane: n = nrow + (lane&15); lane<16: K runs [kb..kb+7],[kb+16..kb+23]; lane>=16: +8
__device__ __forceinline__ v16bf load_bfrag(const __bf16* base, int ld, int nrow, int kbase, int lane) {
  const __bf16* p = base + (nrow + (lane & 15)) * ld + kbase + ((lane & 16) ? 8 : 0);
  v8bf lo = *reinterpret_cast<const v8bf*>(p);
  v8bf hi = *reinterpret_cast<const v8bf*>(p + 16);
  v16bf b;
#pragma unroll
  for (int e = 0; e < 8; ++e) { b[e] = lo[e]; b[8 + e] = hi[e]; }
  return b;
}

__global__ __launch_bounds__(256)
void MultiheadAttention_13365938225757_kernel(const float* __restrict__ Q,
                                              const float* __restrict__ K,
                                              const float* __restrict__ V,
                                              const float* __restrict__ Msk,
                                              float* __restrict__ O) {
  __shared__ __bf16 sQ[BQ * SQ_LD];      // 33280 B
  __shared__ __bf16 sKV[128 * SVT_LD];   // 18432 B: K chunk 64x136 / V^T chunk 128x72
  __shared__ float  sS[BQ * SS_LD];      //  8448 B (cols 64/65 = partial-max scratch)
  __shared__ __bf16 sP[BQ * SP_LD];      //  4608 B
  __shared__ float  sFac[BQ];
  __shared__ float  sM[BQ];
  __shared__ float  sL[BQ];
  __shared__ float  sRed2[64];

  const int tid  = threadIdx.x;
  const int lane = tid & 31;
  const int wave = tid >> 5;
  const int b    = blockIdx.y;
  const int t0   = blockIdx.x * BQ;

  const float* Qp = Q   + ((size_t)b * Td + t0) * Ad;
  const float* Kp = K   + (size_t)b * Sd * Ad;
  const float* Vp = V   + (size_t)b * Sd * Ad;
  const float* Mp = Msk + ((size_t)b * Td + t0) * Sd;

  // ---- stage Q block (32 x 512 fp32 -> bf16 LDS), init row state ----
  if (tid < BQ) { sM[tid] = -1e30f; sL[tid] = 0.0f; }
  for (int i = tid; i < BQ * (Ad / 4); i += 256) {
    const int r  = i >> 7;
    const int c4 = (i & 127) << 2;
    float4 q = *reinterpret_cast<const float4*>(Qp + (size_t)r * Ad + c4);
    v4bf h; h[0] = (__bf16)q.x; h[1] = (__bf16)q.y; h[2] = (__bf16)q.z; h[3] = (__bf16)q.w;
    *reinterpret_cast<v4bf*>(&sQ[r * SQ_LD + c4]) = h;
  }

  const int rt = wave >> 2;   // score row tile (0..1)
  const int ct = wave & 3;    // score col tile (0..3)
  // output col tile co -> global column 16*wave + 128*co (interleaved across waves)

  v8f acc[2][4] = {};

  for (int s0 = 0; s0 < Sd; s0 += BS) {
    // ================= S = Q K^T (A chunked by 128) =================
    v8f sc = {};
#pragma unroll
    for (int a0 = 0; a0 < Ad; a0 += AC) {
      __syncthreads();
      { // prefetch next K chunk (or first V chunk) toward L2
        const int pr = tid >> 2, ps = (tid & 3) * 32;
        if (a0 + AC < Ad) __builtin_prefetch(Kp + (size_t)(s0 + pr) * Ad + (a0 + AC) + ps, 0, 1);
        else              __builtin_prefetch(Vp + (size_t)(s0 + pr) * Ad + ps, 0, 1);
      }
      for (int i = tid; i < BS * (AC / 4); i += 256) {
        const int r  = i >> 5;
        const int c4 = (i & 31) << 2;
        float4 k = *reinterpret_cast<const float4*>(Kp + (size_t)(s0 + r) * Ad + a0 + c4);
        v4bf h; h[0] = (__bf16)k.x; h[1] = (__bf16)k.y; h[2] = (__bf16)k.z; h[3] = (__bf16)k.w;
        *reinterpret_cast<v4bf*>(&sKV[r * SK_LD + c4]) = h;
      }
      __syncthreads();
#pragma unroll
      for (int kk = 0; kk < AC; kk += 32) {
        v16bf af = load_afrag(sQ, SQ_LD, 16 * rt, a0 + kk, lane);
        v16bf bf = load_bfrag(sKV, SK_LD, 16 * ct, kk, lane);
        sc = WMMA_BF16(af, bf, sc);
      }
    }
    { // park scores (C/D layout: row = v + 8*(lane>=16))
      const int rbase = 16 * rt + ((lane & 16) ? 8 : 0);
      const int col   = 16 * ct + (lane & 15);
#pragma unroll
      for (int v = 0; v < 8; ++v) sS[(rbase + v) * SS_LD + col] = sc[v];
    }
    __syncthreads();

    // ---- fold scale + additive mask into sS (coalesced float4) ----
    for (int i = tid; i < BQ * (BS / 4); i += 256) {
      const int r  = i >> 4;
      const int c4 = (i & 15) << 2;
      float4 mk = *reinterpret_cast<const float4*>(Mp + (size_t)r * Sd + s0 + c4);
      float* ps = &sS[r * SS_LD + c4];
      ps[0] = ps[0] * kScale + mk.x;
      ps[1] = ps[1] * kScale + mk.y;
      ps[2] = ps[2] * kScale + mk.z;
      ps[3] = ps[3] * kScale + mk.w;
    }
    __syncthreads();

    // ---- online softmax: 2 threads per row; partial maxes in sS cols 64/65 ----
    if (tid < 64) {
      const int row = tid & 31, half = tid >> 5;
      const float* ps = &sS[row * SS_LD + 32 * half];
      float pm = -1e30f;
      for (int c = 0; c < 32; ++c) pm = fmaxf(pm, ps[c]);
      sS[row * SS_LD + 64 + half] = pm;
    }
    __syncthreads();
    if (tid < 64) {
      const int row = tid & 31, half = tid >> 5;
      const float rowmax = fmaxf(sM[row], fmaxf(sS[row * SS_LD + 64], sS[row * SS_LD + 65]));
      const float* ps = &sS[row * SS_LD + 32 * half];
      __bf16* pp = &sP[row * SP_LD + 32 * half];
      float psum = 0.0f;
      for (int c = 0; c < 32; ++c) {
        float p = __expf(ps[c] - rowmax);
        psum += p;
        pp[c] = (__bf16)p;
      }
      sRed2[tid] = psum;
    }
    __syncthreads();
    if (tid < BQ) {
      const float mo = sM[tid];
      const float rowmax = fmaxf(mo, fmaxf(sS[tid * SS_LD + 64], sS[tid * SS_LD + 65]));
      const float fac = __expf(mo - rowmax);
      sL[tid] = sL[tid] * fac + sRed2[tid] + sRed2[tid + 32];
      sM[tid] = rowmax;
      sFac[tid] = fac;
    }
    __syncthreads();

    // ---- rescale accumulators ----
#pragma unroll
    for (int ro = 0; ro < 2; ++ro) {
      const int rbase = 16 * ro + ((lane & 16) ? 8 : 0);
#pragma unroll
      for (int v = 0; v < 8; ++v) {
        const float f = sFac[rbase + v];
#pragma unroll
        for (int co = 0; co < 4; ++co) acc[ro][co][v] *= f;
      }
    }

    // ---- O += P V : V chunks (128 cols) transposed into LDS, all waves active ----
    v16bf pf[2][2];
#pragma unroll
    for (int ro = 0; ro < 2; ++ro)
#pragma unroll
      for (int k2 = 0; k2 < 2; ++k2)
        pf[ro][k2] = load_afrag(sP, SP_LD, 16 * ro, 32 * k2, lane);

#pragma unroll
    for (int co = 0; co < 4; ++co) {
      __syncthreads();
      { // prefetch next V chunk (or next block's first K chunk)
        const int pr = tid >> 2, ps = (tid & 3) * 32;
        if (co < 3)              __builtin_prefetch(Vp + (size_t)(s0 + pr) * Ad + AC * (co + 1) + ps, 0, 1);
        else if (s0 + BS < Sd)   __builtin_prefetch(Kp + (size_t)(s0 + BS + pr) * Ad + ps, 0, 1);
      }
      // fill V^T chunk: each thread transposes a 4x4 fp32 sub-tile in registers
#pragma unroll
      for (int it = 0; it < 2; ++it) {
        const int idx  = tid + 256 * it;
        const int nsub = (idx & 31) << 2;   // local col base (0..124)
        const int ksub = (idx >> 5) << 2;   // local row base (0..60)
        const float* vb = Vp + (size_t)(s0 + ksub) * Ad + AC * co + nsub;
        float4 q0 = *reinterpret_cast<const float4*>(vb);
        float4 q1 = *reinterpret_cast<const float4*>(vb + Ad);
        float4 q2 = *reinterpret_cast<const float4*>(vb + 2 * Ad);
        float4 q3 = *reinterpret_cast<const float4*>(vb + 3 * Ad);
        __bf16* outp = &sKV[nsub * SVT_LD + ksub];
        v4bf h;
        h[0] = (__bf16)q0.x; h[1] = (__bf16)q1.x; h[2] = (__bf16)q2.x; h[3] = (__bf16)q3.x;
        *reinterpret_cast<v4bf*>(outp) = h;
        h[0] = (__bf16)q0.y; h[1] = (__bf16)q1.y; h[2] = (__bf16)q2.y; h[3] = (__bf16)q3.y;
        *reinterpret_cast<v4bf*>(outp + SVT_LD) = h;
        h[0] = (__bf16)q0.z; h[1] = (__bf16)q1.z; h[2] = (__bf16)q2.z; h[3] = (__bf16)q3.z;
        *reinterpret_cast<v4bf*>(outp + 2 * SVT_LD) = h;
        h[0] = (__bf16)q0.w; h[1] = (__bf16)q1.w; h[2] = (__bf16)q2.w; h[3] = (__bf16)q3.w;
        *reinterpret_cast<v4bf*>(outp + 3 * SVT_LD) = h;
      }
      __syncthreads();
#pragma unroll
      for (int k2 = 0; k2 < 2; ++k2) {
        v16bf bf = load_bfrag(sKV, SVT_LD, 16 * wave, 32 * k2, lane);
#pragma unroll
        for (int ro = 0; ro < 2; ++ro)
          acc[ro][co] = WMMA_BF16(pf[ro][k2], bf, acc[ro][co]);
      }
    }
  }

  // ---- finalize: O /= l, write fp32 ----
  __syncthreads();
  if (tid < BQ) sFac[tid] = 1.0f / sL[tid];
  __syncthreads();

  float* Op = O + ((size_t)b * Td + t0) * Ad;
#pragma unroll
  for (int ro = 0; ro < 2; ++ro) {
    const int rbase = 16 * ro + ((lane & 16) ? 8 : 0);
#pragma unroll
    for (int v = 0; v < 8; ++v) {
      const int row = rbase + v;
      const float f = sFac[row];
#pragma unroll
      for (int co = 0; co < 4; ++co) {
        Op[(size_t)row * Ad + 16 * wave + 128 * co + (lane & 15)] = acc[ro][co][v] * f;
      }
    }
  }
}

extern "C" void kernel_launch(void* const* d_in, const int* in_sizes, int n_in,
                              void* d_out, int out_size, void* d_ws, size_t ws_size,
                              hipStream_t stream) {
  const float* Q   = (const float*)d_in[0];
  const float* K   = (const float*)d_in[1];
  const float* V   = (const float*)d_in[2];
  const float* Msk = (const float*)d_in[3];
  float* O = (float*)d_out;

  const int Bn = in_sizes[0] / (Td * Ad);   // batch count (32)
  dim3 grid(Td / BQ, Bn);
  MultiheadAttention_13365938225757_kernel<<<grid, 256, 0, stream>>>(Q, K, V, Msk, O);
}
